// MOLELinear_51625506897924
// MI455X (gfx1250) — matrix operand order repaired and checked by reading, not compiled, not run on hardware
//
#include <hip/hip_runtime.h>

#define N_ATOMS 65536
#define DIN 512
#define DOUT 512
#define NE 8
#define NS 64
#define BM 128
#define BN 64
#define BK 32
#define MAXT (N_ATOMS / BM + NS) /* 576: upper bound on (system, row-tile) pairs */

#ifndef __has_builtin
#define __has_builtin(x) 0
#endif

#if __has_builtin(__builtin_amdgcn_global_load_async_to_lds_b128)
#define HAS_ASYNC 1
#else
#define HAS_ASYNC 0
#endif

typedef __attribute__((ext_vector_type(16))) __bf16 v16bf;
typedef __attribute__((ext_vector_type(8)))  float  v8f;
typedef __attribute__((ext_vector_type(4)))  int    v4i;

union Frag16 {
    v16bf v;
    uint4 q[2];
};

__device__ __forceinline__ unsigned short f2bf(float f) {
    unsigned u = __float_as_uint(f);
    u += 0x7FFFu + ((u >> 16) & 1u);   // round-to-nearest-even
    return (unsigned short)(u >> 16);
}
__device__ __forceinline__ unsigned pack2(float a, float b) {
    return (unsigned)f2bf(a) | ((unsigned)f2bf(b) << 16);
}

template <int N>
__device__ __forceinline__ void wait_async() {
#if __has_builtin(__builtin_amdgcn_s_wait_asynccnt)
    __builtin_amdgcn_s_wait_asynccnt(N);
#else
    asm volatile("s_wait_asynccnt %0" ::"n"(N) : "memory");
#endif
}

#if HAS_ASYNC
__device__ __forceinline__ void async_cp16(const void* g, void* l) {
    __builtin_amdgcn_global_load_async_to_lds_b128(
        (__attribute__((address_space(1))) v4i*)g,
        (__attribute__((address_space(3))) v4i*)l, 0, 0);
}
#endif

// ---------------------------------------------------------------------------
// Phase A0: Xbf[n,i] = bf16(x[n,i]) — one-time conversion, bandwidth bound.
// ---------------------------------------------------------------------------
__global__ __launch_bounds__(256) void cvt_x_kernel(const float* __restrict__ x,
                                                    unsigned short* __restrict__ Xbf) {
    const int idx = blockIdx.x * 256 + threadIdx.x;   // over groups of 8 floats
    const float4* xp = (const float4*)x + (size_t)idx * 2;
    float4 a = xp[0], b = xp[1];
    uint4 p;
    p.x = pack2(a.x, a.y); p.y = pack2(a.z, a.w);
    p.z = pack2(b.x, b.y); p.w = pack2(b.z, b.w);
    ((uint4*)Xbf)[idx] = p;
}

// ---------------------------------------------------------------------------
// Phase A1: Wmix[s,o,i] = sum_e coeff[s,e] * W[e,o,i], stored as bf16.
// ---------------------------------------------------------------------------
__global__ __launch_bounds__(256) void mix_weights_kernel(
    const float* __restrict__ W, const float* __restrict__ coeff,
    unsigned short* __restrict__ Wmix) {
    const int s = blockIdx.y;
    const int f = blockIdx.x * 256 + threadIdx.x;   // index over float4s of [DOUT*DIN]
    float c[NE];
#pragma unroll
    for (int e = 0; e < NE; e++) c[e] = coeff[s * NE + e];
    const float4* Wp = (const float4*)W;
    float4 acc = make_float4(0.f, 0.f, 0.f, 0.f);
#pragma unroll
    for (int e = 0; e < NE; e++) {
        float4 w = Wp[(size_t)e * (DOUT * DIN / 4) + f];
        acc.x += c[e] * w.x; acc.y += c[e] * w.y;
        acc.z += c[e] * w.z; acc.w += c[e] * w.w;
    }
    uint2 p;
    p.x = pack2(acc.x, acc.y);
    p.y = pack2(acc.z, acc.w);
    ((uint2*)Wmix)[(size_t)s * (DOUT * DIN / 4) + f] = p;
}

// ---------------------------------------------------------------------------
// Phase B: counting sort of batch_indices -> perm grouped by system, plus
// enumeration of (system, row0, rows) GEMM tiles.
// ---------------------------------------------------------------------------
__global__ __launch_bounds__(256) void hist_kernel(const int* __restrict__ bidx,
                                                   int* __restrict__ counts) {
    int n = blockIdx.x * 256 + threadIdx.x;
    atomicAdd(&counts[bidx[n]], 1);
}

__global__ void scan_tiles_kernel(const int* __restrict__ counts,
                                  int* __restrict__ cursor,
                                  int* __restrict__ tileSys,
                                  int* __restrict__ tileRow0,
                                  int* __restrict__ tileRows,
                                  int* __restrict__ numTiles) {
    if (threadIdx.x == 0 && blockIdx.x == 0) {
        int off = 0, t = 0;
        for (int s = 0; s < NS; s++) {
            int cnt = counts[s];
            cursor[s] = off;
            for (int r = 0; r < cnt; r += BM) {
                tileSys[t]  = s;
                tileRow0[t] = off + r;
                tileRows[t] = (cnt - r < BM) ? (cnt - r) : BM;
                t++;
            }
            off += cnt;
        }
        *numTiles = t;
    }
}

__global__ __launch_bounds__(256) void scatter_kernel(const int* __restrict__ bidx,
                                                      int* __restrict__ cursor,
                                                      int* __restrict__ perm) {
    int n = blockIdx.x * 256 + threadIdx.x;
    int pos = atomicAdd(&cursor[bidx[n]], 1);
    perm[pos] = n;   // intra-system order nondeterministic; per-atom result invariant
}

// ---------------------------------------------------------------------------
// Phase C: segmented GEMM. Block = (128-row tile of one system) x (64 cols).
// 256 threads = 8 waves; wave computes a 32x32 patch as 2x2 WMMA 16x16 tiles.
// A/B staged to LDS via async global->LDS B128 copies, double-buffered.
// ---------------------------------------------------------------------------
__global__ __launch_bounds__(256) void moe_gemm_kernel(
    const unsigned short* __restrict__ Xbf, const unsigned short* __restrict__ Wmix,
    const int* __restrict__ perm, const int* __restrict__ tileSys,
    const int* __restrict__ tileRow0, const int* __restrict__ tileRows,
    const int* __restrict__ numTiles, const float* __restrict__ bias,
    float* __restrict__ out) {
    const int bt = blockIdx.x;
    if (bt >= *numTiles) return;
    const int s    = tileSys[bt];
    const int m0   = tileRow0[bt];
    const int rows = tileRows[bt];
    const int n0g  = blockIdx.y * BN;

#if HAS_ASYNC
    __shared__ __align__(16) unsigned As[2][BM * 16];  // bf16 pairs along K
    __shared__ __align__(16) unsigned Bs[2][BN * 16];
#else
    __shared__ __align__(16) unsigned As[1][BM * 16];
    __shared__ __align__(16) unsigned Bs[1][BN * 16];
#endif
    __shared__ int pr[BM];

    const int tid = threadIdx.x;
    if (tid < BM) pr[tid] = perm[m0 + ((tid < rows) ? tid : 0)];
    __syncthreads();

    // Staging map: 16B chunk per (row, quarter). A: rows tid/4 and tid/4+64.
    const int crow = tid >> 2;   // 0..63
    const int cq   = tid & 3;    // 16-byte quarter of a 64B k-slab
    const unsigned short* ga0 = Xbf + (size_t)pr[crow] * DIN + cq * 8;
    const unsigned short* ga1 = Xbf + (size_t)pr[crow + 64] * DIN + cq * 8;
    const unsigned short* gb  =
        Wmix + (size_t)s * DOUT * DIN + (size_t)(n0g + crow) * DIN + cq * 8;
    const int lA0 = crow * 16 + cq * 4;          // uint index in As[buf]
    const int lA1 = (crow + 64) * 16 + cq * 4;
    const int lB  = crow * 16 + cq * 4;

    const int lane = tid & 31;
    const int wave = tid >> 5;
    const int mi = wave & 3;   // 4 row-groups of 32
    const int ni = wave >> 2;  // 2 col-groups of 32
    const bool hi = lane >= 16;
    const int l16 = lane & 15;

    const v8f vz = {0.f, 0.f, 0.f, 0.f, 0.f, 0.f, 0.f, 0.f};
    v8f acc[2][2];
#pragma unroll
    for (int i = 0; i < 2; i++)
#pragma unroll
        for (int j = 0; j < 2; j++) acc[i][j] = vz;

    auto compute_step = [&](const uint4* As4, const uint4* Bs4) {
        Frag16 a[2], b[2];
#pragma unroll
        for (int t = 0; t < 2; t++) {
            // A 16x32 bf16: lanes 0-15 hold K{0-7,16-23}, lanes 16-31 K{8-15,24-31}
            int m = mi * 32 + t * 16 + l16;
            a[t].q[0] = As4[m * 4 + (hi ? 1 : 0)];
            a[t].q[1] = As4[m * 4 + (hi ? 3 : 2)];
            // B 32x16 bf16: N = lane%16; lanes 0-15 hold K{0-7,16-23}, lanes 16-31 K{8-15,24-31}
            int nn = ni * 32 + t * 16 + l16;
            b[t].q[0] = Bs4[nn * 4 + (hi ? 1 : 0)];
            b[t].q[1] = Bs4[nn * 4 + (hi ? 3 : 2)];
        }
#pragma unroll
        for (int i = 0; i < 2; i++)
#pragma unroll
            for (int j = 0; j < 2; j++)
                acc[i][j] = __builtin_amdgcn_wmma_f32_16x16x32_bf16(
                    false, a[i].v, false, b[j].v, (short)0, acc[i][j], false, false);
    };

#if HAS_ASYNC
    auto issue = [&](int buf, int kb) {
        const int ko = kb * BK;   // bf16 units
        async_cp16(ga0 + ko, &As[buf][lA0]);
        async_cp16(ga1 + ko, &As[buf][lA1]);
        async_cp16(gb  + ko, &Bs[buf][lB]);
    };
    issue(0, 0);
    for (int kb = 0; kb < DIN / BK; kb++) {
        const int cur = kb & 1;
        if (kb + 1 < DIN / BK) {
            issue(cur ^ 1, kb + 1);   // prefetch next slab into other buffer
            wait_async<3>();          // older batch (3 asyncs) complete, in order
        } else {
            wait_async<0>();
        }
        __syncthreads();              // all waves' slabs visible
        compute_step((const uint4*)As[cur], (const uint4*)Bs[cur]);
        __syncthreads();              // release buffer 'cur' for overwrite
    }
#else
    for (int kb = 0; kb < DIN / BK; kb++) {
        const int ko = kb * BK;
        *(uint4*)&As[0][lA0] = *(const uint4*)(ga0 + ko);
        *(uint4*)&As[0][lA1] = *(const uint4*)(ga1 + ko);
        *(uint4*)&Bs[0][lB]  = *(const uint4*)(gb + ko);
        __syncthreads();
        compute_step((const uint4*)As[0], (const uint4*)Bs[0]);
        __syncthreads();
    }
#endif

    // Epilogue: C/D 16x16 f32 layout: row = vgpr + (lane>=16 ? 8 : 0), col = lane%16
    const int rbase = hi ? 8 : 0;
#pragma unroll
    for (int i = 0; i < 2; i++) {
#pragma unroll
        for (int j = 0; j < 2; j++) {
            int gcol = n0g + ni * 32 + j * 16 + l16;
            float bv = bias[gcol];
#pragma unroll
            for (int r = 0; r < 8; r++) {
                int lrow = mi * 32 + i * 16 + rbase + r;
                if (lrow < rows) {
                    out[(size_t)pr[lrow] * DOUT + gcol] = acc[i][j][r] + bv;
                }
            }
        }
    }
}

// ---------------------------------------------------------------------------
extern "C" void kernel_launch(void* const* d_in, const int* in_sizes, int n_in,
                              void* d_out, int out_size, void* d_ws, size_t ws_size,
                              hipStream_t stream) {
    (void)in_sizes; (void)n_in; (void)out_size; (void)ws_size;
    const float* x     = (const float*)d_in[0];
    const float* coeff = (const float*)d_in[1];
    const int*   bidx  = (const int*)d_in[2];
    const float* W     = (const float*)d_in[3];
    const float* bias  = (const float*)d_in[4];
    float* out = (float*)d_out;

    char* ws = (char*)d_ws;
    size_t off = 0;
    unsigned short* Wmix = (unsigned short*)(ws + off); off += (size_t)NS * DOUT * DIN * 2;
    unsigned short* Xbf  = (unsigned short*)(ws + off); off += (size_t)N_ATOMS * DIN * 2;
    int* perm     = (int*)(ws + off); off += (size_t)N_ATOMS * 4;
    int* counts   = (int*)(ws + off); off += NS * 4;
    int* cursor   = (int*)(ws + off); off += NS * 4;
    int* tileSys  = (int*)(ws + off); off += MAXT * 4;
    int* tileRow0 = (int*)(ws + off); off += MAXT * 4;
    int* tileRows = (int*)(ws + off); off += MAXT * 4;
    int* numTiles = (int*)(ws + off); off += 4;

    (void)hipMemsetAsync(counts, 0, NS * 4, stream);
    cvt_x_kernel<<<N_ATOMS * DIN / 8 / 256, 256, 0, stream>>>(x, Xbf);
    mix_weights_kernel<<<dim3(DOUT * DIN / 4 / 256, NS), 256, 0, stream>>>(W, coeff, Wmix);
    hist_kernel<<<N_ATOMS / 256, 256, 0, stream>>>(bidx, counts);
    scan_tiles_kernel<<<1, 1, 0, stream>>>(counts, cursor, tileSys, tileRow0, tileRows, numTiles);
    scatter_kernel<<<N_ATOMS / 256, 256, 0, stream>>>(bidx, cursor, perm);
    moe_gemm_kernel<<<dim3(MAXT, DOUT / BN), 256, 0, stream>>>(
        Xbf, Wmix, perm, tileSys, tileRow0, tileRows, numTiles, bias, out);
}
